// FP8MLP_68951404970077
// MI455X (gfx1250) — 8-bit matrix operand order repaired and checked by reading, compiled, not run on hardware
//
#include <hip/hip_runtime.h>
#include <hip/hip_bf16.h>
#include <stdint.h>

#define TDIM 8192
#define HDIM 2048
#define IDIM 8192
#define LDS_STRIDE 144
#define FP8_MAXV 448.0f

typedef __attribute__((ext_vector_type(16))) int   v16i;
typedef __attribute__((ext_vector_type(8)))  float v8f;

static __device__ __forceinline__ float bf16_to_f32(unsigned v) {
  return __uint_as_float(v << 16);
}
static __device__ __forceinline__ unsigned short f32_to_bf16(float f) {
  unsigned u = __float_as_uint(f);
  u += 0x7fffu + ((u >> 16) & 1u);
  return (unsigned short)(u >> 16);
}
static __device__ __forceinline__ unsigned char f32_to_e4m3(float f) {
  unsigned s = (__float_as_uint(f) >> 24) & 0x80u;
  float a = fminf(fabsf(f), FP8_MAXV);
  if (a < 0.015625f) {
    return (unsigned char)(s | (unsigned)__float2int_rn(a * 512.0f));
  }
  unsigned b    = __float_as_uint(a);
  int      e    = (int)((b >> 23) & 0xffu) - 127;
  unsigned mant = b & 0x7fffffu;
  unsigned r    = mant >> 20;
  unsigned rem  = mant & 0xfffffu;
  r += (rem > 0x80000u) || (rem == 0x80000u && (r & 1u));
  unsigned code = ((unsigned)(e + 7) << 3) + r;
  if (code > 0x7eu) code = 0x7eu;
  return (unsigned char)(s | code);
}
static __device__ __forceinline__ float gelu_tanh(float x) {
  return 0.5f * x * (1.0f + tanhf(0.7978845608028654f * (x + 0.044715f * x * x * x)));
}
static __device__ __forceinline__ v8f v8f_zero() {
  v8f z;
#pragma unroll
  for (int i = 0; i < 8; ++i) z[i] = 0.0f;
  return z;
}

static __device__ __forceinline__ void async_load_b128(void* ldsPtr, const void* gPtr) {
  unsigned lds32 = (unsigned)(unsigned long long)ldsPtr;
  asm volatile("global_load_async_to_lds_b128 %0, %1, off"
               :: "v"(lds32), "v"(gPtr)
               : "memory");
}
#if __has_builtin(__builtin_amdgcn_s_wait_asynccnt)
#define WAIT_ASYNC(n) __builtin_amdgcn_s_wait_asynccnt(n)
#else
#define WAIT_ASYNC(n) asm volatile("s_wait_asynccnt %0" :: "i"(n) : "memory")
#endif

template <int ROWS>
static __device__ __forceinline__ void async_copy_tile(unsigned char* lds,
                                                       const unsigned char* g,
                                                       int row0, int gstride, int k0,
                                                       int tid) {
  const unsigned char* gBase = g + (long)row0 * gstride + k0;
#pragma unroll
  for (int it = 0; it < (ROWS * 8) / 256; ++it) {
    int c = it * 256 + tid;
    int r = c >> 3, cc = (c & 7) << 4;
    async_load_b128(lds + r * LDS_STRIDE + cc, gBase + (long)r * gstride + cc);
  }
}

static __device__ __forceinline__ v16i load_a_frag(const unsigned char* lds,
                                                   int rowBase, int lane) {
  const int m = lane & 15, h = lane >> 4;
  const unsigned char* p = lds + (rowBase + m) * LDS_STRIDE + h * 16;
  union { v16i v; unsigned long long d[8]; } f;
#pragma unroll
  for (int j = 0; j < 8; ++j) {
    int off = ((j >> 1) << 5) + ((j & 1) << 3);
    f.d[j] = *(const unsigned long long*)(p + off);
  }
  return f.v;
}
static __device__ __forceinline__ v16i load_b_frag(const unsigned char* lds,
                                                   int rowBase, int lane) {
  const int n = lane & 15, h = lane >> 4;
  const unsigned char* p = lds + (rowBase + n) * LDS_STRIDE + h * 16;
  union { v16i v; uint4 q[4]; } f;
#pragma unroll
  for (int q = 0; q < 4; ++q) f.q[q] = *(const uint4*)(p + q * 32);
  return f.v;
}

__global__ void fp8mlp_zero_kernel(unsigned* p) {
  if (threadIdx.x < 8) p[threadIdx.x] = 0u;
}

__global__ __launch_bounds__(256) void fp8mlp_amax_f32_kernel(
    const float* __restrict__ src, int n4, unsigned* __restrict__ slot) {
  float m = 0.0f;
  for (int i = blockIdx.x * blockDim.x + threadIdx.x; i < n4;
       i += gridDim.x * blockDim.x) {
    float4 v = ((const float4*)src)[i];
    m = fmaxf(m, fmaxf(fmaxf(fabsf(v.x), fabsf(v.y)),
                       fmaxf(fabsf(v.z), fabsf(v.w))));
  }
#pragma unroll
  for (int off = 16; off > 0; off >>= 1) m = fmaxf(m, __shfl_xor(m, off, 32));
  __shared__ float sm[8];
  if ((threadIdx.x & 31) == 0) sm[threadIdx.x >> 5] = m;
  __syncthreads();
  if (threadIdx.x < 8) {
    float v = sm[threadIdx.x];
    v = fmaxf(v, __shfl_xor(v, 4, 32));
    v = fmaxf(v, __shfl_xor(v, 2, 32));
    v = fmaxf(v, __shfl_xor(v, 1, 32));
    if (threadIdx.x == 0) atomicMax(slot, __float_as_uint(v));
  }
}

__global__ __launch_bounds__(256) void fp8mlp_quant_f32_kernel(
    const float* __restrict__ src, unsigned char* __restrict__ dst,
    const unsigned* __restrict__ amaxBits, int n8) {
  const float scale = FP8_MAXV / fmaxf(__uint_as_float(*amaxBits), 1e-12f);
  for (int i = blockIdx.x * blockDim.x + threadIdx.x; i < n8;
       i += gridDim.x * blockDim.x) {
    const int base = i * 8;
    float4 v0 = *(const float4*)(src + base);
    float4 v1 = *(const float4*)(src + base + 4);
    union { unsigned char c[8]; unsigned long long u; } o;
    o.c[0] = f32_to_e4m3(v0.x * scale); o.c[1] = f32_to_e4m3(v0.y * scale);
    o.c[2] = f32_to_e4m3(v0.z * scale); o.c[3] = f32_to_e4m3(v0.w * scale);
    o.c[4] = f32_to_e4m3(v1.x * scale); o.c[5] = f32_to_e4m3(v1.y * scale);
    o.c[6] = f32_to_e4m3(v1.z * scale); o.c[7] = f32_to_e4m3(v1.w * scale);
    *(unsigned long long*)(dst + base) = o.u;
  }
}

__global__ __launch_bounds__(256) void fp8mlp_gateup_kernel(
    const unsigned char* __restrict__ xq, const unsigned char* __restrict__ gq,
    const unsigned char* __restrict__ uq, const unsigned* __restrict__ amaxBits,
    unsigned short* __restrict__ inter, unsigned* __restrict__ interAmax) {
  __shared__ __align__(16) unsigned char sA[2][128 * LDS_STRIDE];
  __shared__ __align__(16) unsigned char sG[2][64 * LDS_STRIDE];
  __shared__ __align__(16) unsigned char sU[2][64 * LDS_STRIDE];

  const int tid = threadIdx.x, lane = tid & 31, wave = tid >> 5;
  const int wm = wave & 3, wn = wave >> 2;
  const int tileN = blockIdx.x * 64, tileM = blockIdx.y * 128;
  const int NK = HDIM / 128;

  v8f accg[2][2], accu[2][2];
#pragma unroll
  for (int a = 0; a < 2; ++a)
#pragma unroll
    for (int b = 0; b < 2; ++b) { accg[a][b] = v8f_zero(); accu[a][b] = v8f_zero(); }

  auto issue = [&](int buf, int k0) {
    async_copy_tile<128>(sA[buf], xq, tileM, HDIM, k0, tid);
    async_copy_tile<64>(sG[buf], gq, tileN, HDIM, k0, tid);
    async_copy_tile<64>(sU[buf], uq, tileN, HDIM, k0, tid);
  };

  issue(0, 0);
  for (int i = 0; i < NK; ++i) {
    const int cur = i & 1;
    if (i + 1 < NK) { issue(cur ^ 1, (i + 1) * 128); WAIT_ASYNC(8); }
    else            { WAIT_ASYNC(0); }
    __syncthreads();

    const v16i a0 = load_a_frag(sA[cur], wm * 32, lane);
    const v16i a1 = load_a_frag(sA[cur], wm * 32 + 16, lane);
#pragma unroll
    for (int ni = 0; ni < 2; ++ni) {
      v16i bg = load_b_frag(sG[cur], wn * 32 + ni * 16, lane);
      accg[0][ni] = __builtin_amdgcn_wmma_f32_16x16x128_fp8_fp8(
          a0, bg, (short)0, accg[0][ni], false, false);
      accg[1][ni] = __builtin_amdgcn_wmma_f32_16x16x128_fp8_fp8(
          a1, bg, (short)0, accg[1][ni], false, false);
      v16i bu = load_b_frag(sU[cur], wn * 32 + ni * 16, lane);
      accu[0][ni] = __builtin_amdgcn_wmma_f32_16x16x128_fp8_fp8(
          a0, bu, (short)0, accu[0][ni], false, false);
      accu[1][ni] = __builtin_amdgcn_wmma_f32_16x16x128_fp8_fp8(
          a1, bu, (short)0, accu[1][ni], false, false);
    }
    __syncthreads();
  }

  const float ax = fmaxf(__uint_as_float(amaxBits[0]), 1e-12f) * (1.0f / FP8_MAXV);
  const float ag = fmaxf(__uint_as_float(amaxBits[1]), 1e-12f) * (1.0f / FP8_MAXV);
  const float au = fmaxf(__uint_as_float(amaxBits[2]), 1e-12f) * (1.0f / FP8_MAXV);
  const float sg = ax * ag, su = ax * au;

  const int n = lane & 15, h = lane >> 4;
  float lm = 0.0f;
#pragma unroll
  for (int mi = 0; mi < 2; ++mi) {
#pragma unroll
    for (int ni = 0; ni < 2; ++ni) {
#pragma unroll
      for (int r = 0; r < 8; ++r) {
        const long row = tileM + wm * 32 + mi * 16 + r + 8 * h;
        const int  col = tileN + wn * 32 + ni * 16 + n;
        float gv = bf16_to_f32(f32_to_bf16(accg[mi][ni][r] * sg));
        gv = gelu_tanh(gv);
        float uv = bf16_to_f32(f32_to_bf16(accu[mi][ni][r] * su));
        unsigned short pb = f32_to_bf16(gv * uv);
        inter[row * IDIM + col] = pb;
        lm = fmaxf(lm, fabsf(bf16_to_f32(pb)));
      }
    }
  }
#pragma unroll
  for (int off = 16; off > 0; off >>= 1) lm = fmaxf(lm, __shfl_xor(lm, off, 32));
  if (lane == 0) atomicMax(interAmax, __float_as_uint(lm));
}

__global__ __launch_bounds__(256) void fp8mlp_quant_bf16_kernel(
    const unsigned short* __restrict__ src, unsigned char* __restrict__ dst,
    const unsigned* __restrict__ amaxBits, int n8) {
  const float scale = FP8_MAXV / fmaxf(__uint_as_float(*amaxBits), 1e-12f);
  for (int i = blockIdx.x * blockDim.x + threadIdx.x; i < n8;
       i += gridDim.x * blockDim.x) {
    const int base = i * 8;
    uint4 raw = *(const uint4*)(src + base);
    unsigned w[4] = {raw.x, raw.y, raw.z, raw.w};
    union { unsigned char c[8]; unsigned long long u; } o;
#pragma unroll
    for (int j = 0; j < 4; ++j) {
      o.c[2 * j + 0] = f32_to_e4m3(bf16_to_f32(w[j] & 0xffffu) * scale);
      o.c[2 * j + 1] = f32_to_e4m3(bf16_to_f32(w[j] >> 16) * scale);
    }
    *(unsigned long long*)(dst + base) = o.u;
  }
}

__global__ __launch_bounds__(256) void fp8mlp_down_kernel(
    const unsigned char* __restrict__ iq, const unsigned char* __restrict__ dq,
    const unsigned* __restrict__ amaxBits, unsigned short* __restrict__ out) {
  __shared__ __align__(16) unsigned char sA[2][128 * LDS_STRIDE];
  __shared__ __align__(16) unsigned char sB[2][128 * LDS_STRIDE];

  const int tid = threadIdx.x, lane = tid & 31, wave = tid >> 5;
  const int wm = wave & 3, wn = wave >> 2;
  const int tileN = blockIdx.x * 128, tileM = blockIdx.y * 128;
  const int NK = IDIM / 128;

  v8f acc[2][4];
#pragma unroll
  for (int a = 0; a < 2; ++a)
#pragma unroll
    for (int b = 0; b < 4; ++b) acc[a][b] = v8f_zero();

  auto issue = [&](int buf, int k0) {
    async_copy_tile<128>(sA[buf], iq, tileM, IDIM, k0, tid);
    async_copy_tile<128>(sB[buf], dq, tileN, IDIM, k0, tid);
  };

  issue(0, 0);
  for (int i = 0; i < NK; ++i) {
    const int cur = i & 1;
    if (i + 1 < NK) { issue(cur ^ 1, (i + 1) * 128); WAIT_ASYNC(8); }
    else            { WAIT_ASYNC(0); }
    __syncthreads();

    const v16i a0 = load_a_frag(sA[cur], wm * 32, lane);
    const v16i a1 = load_a_frag(sA[cur], wm * 32 + 16, lane);
#pragma unroll
    for (int ni = 0; ni < 4; ++ni) {
      v16i b = load_b_frag(sB[cur], wn * 64 + ni * 16, lane);
      acc[0][ni] = __builtin_amdgcn_wmma_f32_16x16x128_fp8_fp8(
          a0, b, (short)0, acc[0][ni], false, false);
      acc[1][ni] = __builtin_amdgcn_wmma_f32_16x16x128_fp8_fp8(
          a1, b, (short)0, acc[1][ni], false, false);
    }
    __syncthreads();
  }

  const float si = fmaxf(__uint_as_float(amaxBits[4]), 1e-12f) * (1.0f / FP8_MAXV);
  const float sd = fmaxf(__uint_as_float(amaxBits[3]), 1e-12f) * (1.0f / FP8_MAXV);
  const float sc = si * sd;
  const int n = lane & 15, h = lane >> 4;
#pragma unroll
  for (int mi = 0; mi < 2; ++mi) {
#pragma unroll
    for (int ni = 0; ni < 4; ++ni) {
#pragma unroll
      for (int r = 0; r < 8; ++r) {
        const long row = tileM + wm * 32 + mi * 16 + r + 8 * h;
        const int  col = tileN + wn * 64 + ni * 16 + n;
        out[row * HDIM + col] = f32_to_bf16(acc[mi][ni][r] * sc);
      }
    }
  }
}

extern "C" void kernel_launch(void* const* d_in, const int* in_sizes, int n_in,
                              void* d_out, int out_size, void* d_ws, size_t ws_size,
                              hipStream_t stream) {
  (void)in_sizes; (void)n_in; (void)out_size; (void)ws_size;
  const float* x  = (const float*)d_in[0];
  const float* gw = (const float*)d_in[1];
  const float* uw = (const float*)d_in[2];
  const float* dw = (const float*)d_in[3];
  unsigned short* out = (unsigned short*)d_out;

  const size_t TH = (size_t)TDIM * HDIM;
  const size_t TI = (size_t)TDIM * IDIM;

  unsigned char* ws = (unsigned char*)d_ws;
  unsigned*       amax  = (unsigned*)ws;
  unsigned char*  xq    = ws + 256;
  unsigned char*  gq    = xq + TH;
  unsigned char*  uq    = gq + TH;
  unsigned char*  dq    = uq + TH;
  unsigned short* inter = (unsigned short*)(dq + TH);
  unsigned char*  iq    = (unsigned char*)(inter + TI);

  fp8mlp_zero_kernel<<<1, 32, 0, stream>>>(amax);

  fp8mlp_amax_f32_kernel<<<1024, 256, 0, stream>>>(x,  (int)(TH / 4), amax + 0);
  fp8mlp_amax_f32_kernel<<<1024, 256, 0, stream>>>(gw, (int)(TH / 4), amax + 1);
  fp8mlp_amax_f32_kernel<<<1024, 256, 0, stream>>>(uw, (int)(TH / 4), amax + 2);
  fp8mlp_amax_f32_kernel<<<1024, 256, 0, stream>>>(dw, (int)(TH / 4), amax + 3);

  const int n8 = (int)(TH / 8);
  const int qb = (n8 + 255) / 256;
  fp8mlp_quant_f32_kernel<<<qb, 256, 0, stream>>>(x,  xq, amax + 0, n8);
  fp8mlp_quant_f32_kernel<<<qb, 256, 0, stream>>>(gw, gq, amax + 1, n8);
  fp8mlp_quant_f32_kernel<<<qb, 256, 0, stream>>>(uw, uq, amax + 2, n8);
  fp8mlp_quant_f32_kernel<<<qb, 256, 0, stream>>>(dw, dq, amax + 3, n8);

  dim3 g1(IDIM / 64, TDIM / 128);
  fp8mlp_gateup_kernel<<<g1, 256, 0, stream>>>(xq, gq, uq, amax, inter, amax + 4);

  const int n8i = (int)(TI / 8);
  fp8mlp_quant_bf16_kernel<<<(n8i + 255) / 256, 256, 0, stream>>>(inter, iq,
                                                                  amax + 4, n8i);

  dim3 g2(HDIM / 128, TDIM / 128);
  fp8mlp_down_kernel<<<g2, 256, 0, stream>>>(iq, dq, amax, out);
}
